// BahdanauAttention_16020228014217
// MI455X (gfx1250) — matrix operand order repaired and checked by reading, not compiled, not run on hardware
//
#include <hip/hip_runtime.h>
#include <hip/hip_bf16.h>
#include <math.h>

// Problem constants (match reference setup_inputs)
#define S_LEN 2048
#define BATCH 32
#define HDIM  1024
#define ADIM  512

typedef __attribute__((ext_vector_type(16))) __bf16        v16bf;
typedef __attribute__((ext_vector_type(2)))  __bf16        v2bf;
typedef __attribute__((ext_vector_type(8)))  float         v8f;
typedef __attribute__((ext_vector_type(4)))  unsigned int  v4u;
typedef __attribute__((ext_vector_type(2)))  unsigned int  v2u;
typedef __attribute__((ext_vector_type(4)))  float         v4f;

union ABfrag { v16bf v; v4u q[2]; };

__device__ __forceinline__ unsigned short f2bf(float f) {
    unsigned u = __float_as_uint(f);
    unsigned r = (u + 0x7FFFu + ((u >> 16) & 1u)) >> 16;   // round-to-nearest-even
    return (unsigned short)r;
}

// Pack two fp32 -> packed bf16x2 dword. Prefer the HW v_cvt_pk_bf16_f32 if present.
__device__ __forceinline__ unsigned pack_bf16(float x, float y) {
#if __has_builtin(__builtin_amdgcn_cvt_pk_bf16_f32)
    v2bf r = __builtin_amdgcn_cvt_pk_bf16_f32(x, y);
    union { v2bf v; unsigned u; } c; c.v = r;
    return c.u;
#else
    return (unsigned)f2bf(x) | ((unsigned)f2bf(y) << 16);
#endif
}

// ---------------- Kernel 1: W_enc fp32 -> bf16 (stays hot in L2) -------------
__global__ void k_cvt_w(const float* __restrict__ W, unsigned short* __restrict__ Wb, int n) {
    int i = blockIdx.x * blockDim.x + threadIdx.x;
    if (i < n) Wb[i] = f2bf(W[i]);
}

// ---------------- Kernel 2: dec_att[b,a] = dec_out[b,:] . W_dec[a,:] ---------
__global__ void k_dec_att(const float* __restrict__ dec, const float* __restrict__ Wd,
                          float* __restrict__ da) {
    int idx = blockIdx.x * blockDim.x + threadIdx.x;     // 32*512
    int a = idx & (ADIM - 1);
    int b = idx >> 9;
    const float* x = dec + b * HDIM;
    const float* w = Wd + a * HDIM;
    float s = 0.f;
    #pragma unroll 8
    for (int h = 0; h < HDIM; ++h) s += x[h] * w[h];
    da[idx] = s;
}

// ---------------- Kernel 3: fused WMMA GEMM + tanh·att_v -> scores(B,S) ------
// grid = S_LEN blocks (one s per block), block = 256 threads = 8 waves.
// Rows of the GEMM A-matrix are the 32 batch rows at this s (contiguous).
__global__ __launch_bounds__(256)
void k_scores(const float* __restrict__ enc, const unsigned short* __restrict__ Wb,
              const float* __restrict__ dec_att, const float* __restrict__ att_v,
              float* __restrict__ scores) {
    __shared__ unsigned short Atile[2][32 * 40]; // ping-pong 32x32 bf16, padded stride 40
    __shared__ float sPart[4 * 32];              // per-nchunk partial scores

    const int s    = blockIdx.x;
    const int tid  = threadIdx.x;
    const int w    = tid >> 5;
    const int lane = tid & 31;
    const int mi   = w & 1;          // which 16-row tile (b 0..15 / 16..31)
    const int nch  = w >> 1;         // 4 column chunks of 128
    const int n0   = nch * 128;
    const int nlo  = lane & 15;
    const int half = lane >> 4;
    const int arow = mi * 16 + nlo;  // A-fragment row held by this lane

    const size_t rowBase = (size_t)s * BATCH * HDIM;

    v8f acc[8];
    #pragma unroll
    for (int nt = 0; nt < 8; ++nt) acc[nt] = (v8f){0.f,0.f,0.f,0.f,0.f,0.f,0.f,0.f};

    const int ldr = tid >> 3;          // cooperative A-tile load: row 0..31
    const int ldc = (tid & 7) * 4;     // 4 floats per thread
    const float* aptr = enc + rowBase + (size_t)ldr * HDIM + ldc;

    // Per-lane base of this wave's B-fragment column in bf16 W
    const unsigned short* wbase = Wb + (size_t)(n0 + nlo) * HDIM + half * 8;

    // prologue: stage k0 = 0 tile into buffer 0
    {
        v4f f = *(const v4f*)(aptr);
        v2u p;
        p.x = pack_bf16(f.x, f.y);
        p.y = pack_bf16(f.z, f.w);
        *(v2u*)&Atile[0][ldr * 40 + ldc] = p;
    }
    __syncthreads();

    int cur = 0;
    for (int k0 = 0; k0 < HDIM; k0 += 32) {
        const bool have_next = (k0 + 32) < HDIM;   // uniform across block

        // issue next A-tile global load first (retires before the B frags below)
        v4f fn;
        if (have_next) fn = *(const v4f*)(aptr + k0 + 32);

        // A fragment: V0-3 = K kA..kA+7, V4-7 = K kA+16..kA+23, kA = half*8
        ABfrag af;
        af.q[0] = *(const v4u*)&Atile[cur][arow * 40 + half * 8];
        af.q[1] = *(const v4u*)&Atile[cur][arow * 40 + half * 8 + 16];

        // preload ALL 8 B fragments into distinct registers (one big clause),
        // so the 8 WMMAs below overlap with the in-flight loads.
        ABfrag bf_[8];
        #pragma unroll
        for (int nt = 0; nt < 8; ++nt) {
            const unsigned short* wp = wbase + (size_t)(nt * 16) * HDIM + k0;
            bf_[nt].q[0] = *(const v4u*)(wp);
            bf_[nt].q[1] = *(const v4u*)(wp + 16);
        }

        #pragma unroll
        for (int nt = 0; nt < 8; ++nt) {
            acc[nt] = __builtin_amdgcn_wmma_f32_16x16x32_bf16(
                false, af.v, false, bf_[nt].v, (short)0, acc[nt], false, false);
        }

        // convert + stage the prefetched A tile while WMMAs drain
        if (have_next) {
            v2u pn;
            pn.x = pack_bf16(fn.x, fn.y);
            pn.y = pack_bf16(fn.z, fn.w);
            *(v2u*)&Atile[cur ^ 1][ldr * 40 + ldc] = pn;
        }
        __syncthreads();   // single barrier per k-step (ping-pong)
        cur ^= 1;
    }

    // Epilogue: rowsum over this wave's 128 columns of tanh(acc + dec_att)*att_v
    float rowsum[8];
    #pragma unroll
    for (int v = 0; v < 8; ++v) rowsum[v] = 0.f;

    #pragma unroll
    for (int nt = 0; nt < 8; ++nt) {
        int a_col = n0 + nt * 16 + nlo;
        float av = att_v[a_col];
        const float* da = dec_att + a_col;
        #pragma unroll
        for (int v = 0; v < 8; ++v) {
            int b = mi * 16 + half * 8 + v;          // C/D layout: row = v + 8*half
            rowsum[v] += tanhf(acc[nt][v] + da[b * ADIM]) * av;
        }
    }

    // deterministic butterfly reduce across the 16 lanes sharing each row
    #pragma unroll
    for (int m = 1; m < 16; m <<= 1)
        #pragma unroll
        for (int v = 0; v < 8; ++v)
            rowsum[v] += __shfl_xor(rowsum[v], m, 32);

    if (nlo == 0) {
        #pragma unroll
        for (int v = 0; v < 8; ++v)
            sPart[nch * 32 + mi * 16 + half * 8 + v] = rowsum[v];
    }
    __syncthreads();
    if (tid < BATCH) {
        int b = tid;
        float sum = sPart[b] + sPart[32 + b] + sPart[64 + b] + sPart[96 + b];
        scores[b * S_LEN + s] = sum;
    }
}

// ---------------- Kernel 4: softmax over s per batch row ---------------------
__global__ __launch_bounds__(256)
void k_softmax(const float* __restrict__ scores, float* __restrict__ wout) {
    __shared__ float red[256];
    const int b = blockIdx.x, t = threadIdx.x;
    const float* row = scores + b * S_LEN;

    float vals[8], mx = -3.4e38f;
    #pragma unroll
    for (int i = 0; i < 8; ++i) { vals[i] = row[t + i * 256]; mx = fmaxf(mx, vals[i]); }
    red[t] = mx; __syncthreads();
    for (int off = 128; off > 0; off >>= 1) {
        if (t < off) red[t] = fmaxf(red[t], red[t + off]);
        __syncthreads();
    }
    float m = red[0]; __syncthreads();

    float se = 0.f;
    #pragma unroll
    for (int i = 0; i < 8; ++i) { vals[i] = expf(vals[i] - m); se += vals[i]; }
    red[t] = se; __syncthreads();
    for (int off = 128; off > 0; off >>= 1) {
        if (t < off) red[t] += red[t + off];
        __syncthreads();
    }
    float inv = 1.f / red[0];
    #pragma unroll
    for (int i = 0; i < 8; ++i) wout[b * S_LEN + t + i * 256] = vals[i] * inv;
}

// ---------------- Kernel 5: context partial sums over s-chunks ---------------
// grid = 8 s-chunks * 32 b * 4 h-chunks = 1024 blocks, 256 threads (one h each)
__global__ __launch_bounds__(256)
void k_ctx_part(const float* __restrict__ enc, const float* __restrict__ wout,
                float* __restrict__ part) {
    __shared__ float lw[256];
    const int bx = blockIdx.x;
    const int sc = bx & 7;
    const int b  = (bx >> 3) & 31;
    const int hc = bx >> 8;
    const int t  = threadIdx.x;
    const int h  = hc * 256 + t;

    lw[t] = wout[b * S_LEN + sc * 256 + t];
    __syncthreads();

    float acc = 0.f;
    const float* ep = enc + ((size_t)(sc * 256) * BATCH + b) * HDIM + h;
    #pragma unroll 4
    for (int i = 0; i < 256; ++i) { acc += lw[i] * ep[0]; ep += (size_t)BATCH * HDIM; }
    part[(size_t)(sc * 32 + b) * HDIM + h] = acc;
}

// ---------------- Kernel 6: reduce partials -> context -----------------------
__global__ void k_ctx_reduce(const float* __restrict__ part, float* __restrict__ ctx) {
    int i = blockIdx.x * blockDim.x + threadIdx.x;   // 32*1024
    float s = 0.f;
    #pragma unroll
    for (int c = 0; c < 8; ++c) s += part[c * (BATCH * HDIM) + i];
    ctx[i] = s;
}

extern "C" void kernel_launch(void* const* d_in, const int* in_sizes, int n_in,
                              void* d_out, int out_size, void* d_ws, size_t ws_size,
                              hipStream_t stream) {
    const float* dec_out  = (const float*)d_in[0];   // (32,1024)
    const float* enc_outs = (const float*)d_in[1];   // (2048,32,1024)
    const float* W_enc    = (const float*)d_in[2];   // (512,1024)
    const float* W_dec    = (const float*)d_in[3];   // (512,1024)
    const float* att_v    = (const float*)d_in[4];   // (512)

    float* ctx_out = (float*)d_out;                  // (32,1024)
    float* w_out   = (float*)d_out + BATCH * HDIM;   // (32,2048)

    char* ws = (char*)d_ws;
    unsigned short* Wb  = (unsigned short*)ws;                               // 1 MB
    float* dec_att      = (float*)(ws + (size_t)ADIM * HDIM * 2);            // 64 KB
    float* scores       = (float*)((char*)dec_att + (size_t)BATCH * ADIM * 4);   // 256 KB
    float* part         = (float*)((char*)scores + (size_t)BATCH * S_LEN * 4);   // 1 MB

    k_cvt_w     <<<(ADIM * HDIM) / 256, 256, 0, stream>>>(W_enc, Wb, ADIM * HDIM);
    k_dec_att   <<<(BATCH * ADIM) / 256, 256, 0, stream>>>(dec_out, W_dec, dec_att);
    k_scores    <<<S_LEN, 256, 0, stream>>>(enc_outs, Wb, dec_att, att_v, scores);
    k_softmax   <<<BATCH, 256, 0, stream>>>(scores, w_out);
    k_ctx_part  <<<8 * BATCH * 4, 256, 0, stream>>>(enc_outs, w_out, part);
    k_ctx_reduce<<<(BATCH * HDIM) / 256, 256, 0, stream>>>(part, ctx_out);
}
